// LiquidEmberBlock_16655883174425
// MI455X (gfx1250) — matrix operand order repaired and checked by reading, not compiled, not run on hardware
//
#include <hip/hip_runtime.h>
#include <math.h>

typedef __bf16 bf16_t;
typedef __attribute__((ext_vector_type(16))) __bf16 v16bf;
typedef __attribute__((ext_vector_type(8)))  __bf16 v8bf;
typedef __attribute__((ext_vector_type(8)))  float  v8f;

#define B_  8
#define S_  2048
#define E_  768
#define U_  1024
#define BB_ 512
#define HM_ 3072
#define ROWS (B_*S_)   // 16384

// ---------------------------------------------------------------- WMMA helpers
__device__ __forceinline__ v8f wmma_bf16(v16bf a, v16bf b, v8f c) {
  return __builtin_amdgcn_wmma_f32_16x16x32_bf16(false, a, false, b, (short)0, c,
                                                 false, false);
}

// A fragment: 16x32 bf16, row-major A with leading dim lda.
// lanes 0-15: M=lane, elems 0-7 = K[ko..ko+7],  elems 8-15 = K[ko+16..ko+23]
// lanes 16-31: M=lane-16, elems 0-7 = K[ko+8..], elems 8-15 = K[ko+24..]
__device__ __forceinline__ v16bf ldA(const bf16_t* A, int lda, int m0, int ko, int lane) {
  const bf16_t* p = A + (size_t)(m0 + (lane & 15)) * lda + ko + ((lane >> 4) << 3);
  union { v16bf v; v8bf h[2]; } u;
  u.h[0] = *(const v8bf*)(p);
  u.h[1] = *(const v8bf*)(p + 16);
  return u.v;
}

// B fragment: 32x16 bf16 from transposed (N-major) weights Bt[N][K], ldb = K.
// lane holds column n = n0 + lane%16, elems 0-7 = K[ko+(lane/16)*8..+7], elems 8-15 = K[ko+16+(lane/16)*8..+7]
__device__ __forceinline__ v16bf ldB(const bf16_t* Bt, int ldb, int n0, int ko, int lane) {
  const bf16_t* p = Bt + (size_t)(n0 + (lane & 15)) * ldb + ko + ((lane >> 4) << 3);
  union { v16bf v; v8bf h[2]; } u;
  u.h[0] = *(const v8bf*)(p);
  u.h[1] = *(const v8bf*)(p + 16);
  return u.v;
}

// 2x2 register-blocked inner loop: 32x32 output per wave, A/B fragment reuse.
__device__ __forceinline__ void gemm_2x2(const bf16_t* __restrict__ A, int lda,
                                         const bf16_t* __restrict__ Bt, int ldb,
                                         int K, int m0, int n0, int lane, v8f acc[4]) {
  for (int ko = 0; ko < K; ko += 32) {
    v16bf a0 = ldA(A, lda, m0,      ko, lane);
    v16bf a1 = ldA(A, lda, m0 + 16, ko, lane);
    v16bf b0 = ldB(Bt, ldb, n0,      ko, lane);
    v16bf b1 = ldB(Bt, ldb, n0 + 16, ko, lane);
    acc[0] = wmma_bf16(a0, b0, acc[0]);
    acc[1] = wmma_bf16(a0, b1, acc[1]);
    acc[2] = wmma_bf16(a1, b0, acc[2]);
    acc[3] = wmma_bf16(a1, b1, acc[3]);
  }
}

__device__ __forceinline__ float lecun_tanh(float x) { return 1.7159f * tanhf(0.666f * x); }
__device__ __forceinline__ float sigmoidf_(float x)  { return 1.f / (1.f + expf(-x)); }

// ------------------------------------------------- weight convert + transpose
// src: f32 [K][N] row-major  ->  dst: bf16 [N][K]
__global__ void k_transpose(const float* __restrict__ src, bf16_t* __restrict__ dst,
                            int K, int N) {
  int idx = blockIdx.x * blockDim.x + threadIdx.x;
  if (idx < K * N) {
    int n = idx / K, k = idx % K;
    dst[idx] = (bf16_t)src[(size_t)k * N + n];
  }
}

// ------------------------------------------------------------------ LayerNorm1
__global__ __launch_bounds__(256) void k_ln1(const float* __restrict__ x,
                                             const float* __restrict__ g,
                                             const float* __restrict__ b,
                                             bf16_t* __restrict__ out) {
  int row = blockIdx.x;
  const float* xr = x + (size_t)row * E_;
  float v[3], s = 0.f, s2 = 0.f;
#pragma unroll
  for (int i = 0; i < 3; ++i) {
    v[i] = xr[threadIdx.x + i * 256];
    s += v[i]; s2 += v[i] * v[i];
  }
  __shared__ float red[16];
  for (int off = 16; off > 0; off >>= 1) {
    s  += __shfl_down(s,  off, 32);
    s2 += __shfl_down(s2, off, 32);
  }
  if ((threadIdx.x & 31) == 0) { red[(threadIdx.x >> 5) * 2] = s; red[(threadIdx.x >> 5) * 2 + 1] = s2; }
  __syncthreads();
  if (threadIdx.x == 0) {
    float a = 0.f, c = 0.f;
    for (int w = 0; w < 8; ++w) { a += red[2 * w]; c += red[2 * w + 1]; }
    red[0] = a / (float)E_; red[1] = c / (float)E_;
  }
  __syncthreads();
  float mu = red[0];
  float rs = rsqrtf(red[1] - mu * mu + 1e-5f);
#pragma unroll
  for (int i = 0; i < 3; ++i) {
    int c = threadIdx.x + i * 256;
    out[(size_t)row * E_ + c] = (bf16_t)((v[i] - mu) * rs * g[c] + b[c]);
  }
}

// ------------------------- bbx = normed @ Wbb[:E] + b_bb   (f32 out, parallel)
// macro-tiles: 512 (M) x 16 (N) of 32x32 each -> 8192 wave jobs, 1024 blocks
__global__ __launch_bounds__(256) void k_gemm_bbx(const bf16_t* __restrict__ A,
                                                  const bf16_t* __restrict__ Bt,
                                                  const float* __restrict__ bias,
                                                  float* __restrict__ C) {
  int wid = blockIdx.x * 8 + (threadIdx.x >> 5), lane = threadIdx.x & 31;
  int mt = wid >> 4, nt = wid & 15;
  int m0 = mt * 32, n0 = nt * 32;
  v8f acc[4] = {};
  gemm_2x2(A, E_, Bt, E_, E_, m0, n0, lane, acc);
#pragma unroll
  for (int j = 0; j < 2; ++j) {
    int n = n0 + j * 16 + (lane & 15);
    float bv = bias[n];
#pragma unroll
    for (int i = 0; i < 2; ++i) {
      int mb = m0 + i * 16 + ((lane >> 4) << 3);
      v8f a = acc[i * 2 + j];
#pragma unroll
      for (int r = 0; r < 8; ++r)
        C[(size_t)(mb + r) * BB_ + n] = a[r] + bv;
    }
  }
}

// ------------------------------------------------ persistent CfC scan (1 WGP)
__global__ __launch_bounds__(512) void k_scan(const float* __restrict__ hx,
                                              const float* __restrict__ bbx,
                                              const bf16_t* __restrict__ WbbhT,
                                              const bf16_t* __restrict__ WheadsT,
                                              const float* __restrict__ bff1,
                                              const float* __restrict__ bff2,
                                              const float* __restrict__ bta,
                                              const float* __restrict__ btb,
                                              bf16_t* __restrict__ cfc,
                                              float* __restrict__ hx_out) {
  __shared__ __align__(32) bf16_t hbf[16 * U_];   // padded-M h state (rows 8-15 zero)
  __shared__ __align__(32) bf16_t bbs[16 * BB_];  // backbone activations
  int tid = threadIdx.x, lane = tid & 31, wv = tid >> 5;  // 16 waves

  for (int i = tid; i < 16 * U_; i += 512)
    hbf[i] = (i < B_ * U_) ? (bf16_t)hx[i] : (bf16_t)0.0f;
  for (int i = tid; i < 16 * BB_; i += 512) bbs[i] = (bf16_t)0.0f;
  __syncthreads();

  for (int t = 0; t < S_; ++t) {
    // GEMM1: bbs = lecun_tanh(hbf @ WbbhT^T + bbx[t])
    // one K-loop per wave covering both of its N-tiles (shared A fragment)
    {
      int n0 = wv * 32;
      v8f acc0 = {}, acc1 = {};
      for (int ko = 0; ko < U_; ko += 32) {
        v16bf a = ldA(hbf, U_, 0, ko, lane);
        acc0 = wmma_bf16(a, ldB(WbbhT, U_, n0,      ko, lane), acc0);
        acc1 = wmma_bf16(a, ldB(WbbhT, U_, n0 + 16, ko, lane), acc1);
      }
      if (lane < 16) {
#pragma unroll
        for (int j = 0; j < 2; ++j) {
          int n = n0 + j * 16 + lane;
          v8f a = j ? acc1 : acc0;
#pragma unroll
          for (int r = 0; r < 8; ++r) {   // valid batch rows M=r (lanes 0-15)
            float v = a[r] + bbx[(size_t)(r * S_ + t) * BB_ + n];
            bbs[r * BB_ + n] = (bf16_t)lecun_tanh(v);
          }
        }
      }
    }
    __syncthreads();
    // GEMM2: all four heads for the same 16 columns in one wave -> in-register combine
    for (int c = 0; c < 4; ++c) {
      int n0 = (wv * 4 + c) * 16;       // columns n0..n0+15 of U
      v8f a1 = {}, a2 = {}, a3 = {}, a4 = {};
      for (int ko = 0; ko < BB_; ko += 32) {
        v16bf af = ldA(bbs, BB_, 0, ko, lane);
        a1 = wmma_bf16(af, ldB(WheadsT + 0 * (U_ * BB_), BB_, n0, ko, lane), a1);
        a2 = wmma_bf16(af, ldB(WheadsT + 1 * (U_ * BB_), BB_, n0, ko, lane), a2);
        a3 = wmma_bf16(af, ldB(WheadsT + 2 * (U_ * BB_), BB_, n0, ko, lane), a3);
        a4 = wmma_bf16(af, ldB(WheadsT + 3 * (U_ * BB_), BB_, n0, ko, lane), a4);
      }
      if (lane < 16) {
        int u = n0 + lane;
        float B1 = bff1[u], B2 = bff2[u], B3 = bta[u], B4 = btb[u];
#pragma unroll
        for (int r = 0; r < 8; ++r) {
          float f1 = lecun_tanh(a1[r] + B1);
          float f2 = lecun_tanh(a2[r] + B2);
          float ti = sigmoidf_(a3[r] + B3 + a4[r] + B4);
          float hn = f1 * (1.f - ti) + ti * f2;
          hbf[(r << 10) + u] = (bf16_t)hn;
          if (u < E_) cfc[(size_t)(r * S_ + t) * E_ + u] = (bf16_t)hn;
        }
      }
    }
    __syncthreads();
  }
  for (int i = tid; i < B_ * U_; i += 512) hx_out[i] = (float)hbf[i];
}

// -------------------- LIF gate + residual (-> d_out as x1) + LayerNorm2 (-> h2)
__global__ __launch_bounds__(256) void k_post(const float* __restrict__ x,
                                              const bf16_t* __restrict__ cfc,
                                              const float* __restrict__ thr,
                                              const float* __restrict__ leak,
                                              const float* __restrict__ steep,
                                              const float* __restrict__ g,
                                              const float* __restrict__ bb_,
                                              float* __restrict__ xout,
                                              bf16_t* __restrict__ h2) {
  int row = blockIdx.x;
  float v[3], s = 0.f, s2 = 0.f;
#pragma unroll
  for (int i = 0; i < 3; ++i) {
    int c = threadIdx.x + i * 256;
    float xv = x[(size_t)row * E_ + c];
    float cv = (float)cfc[(size_t)row * E_ + c];
    float th = fabsf(thr[c]) * 0.1f;
    float ls = sigmoidf_(leak[c]);
    float st = log1pf(expf(steep[c]));
    float fire = sigmoidf_(st * (fabsf(cv) - th));
    float gate = fire + ls * (1.f - fire);
    float x1 = xv + cv * gate;
    xout[(size_t)row * E_ + c] = x1;
    v[i] = x1; s += x1; s2 += x1 * x1;
  }
  __shared__ float red[16];
  for (int off = 16; off > 0; off >>= 1) {
    s  += __shfl_down(s,  off, 32);
    s2 += __shfl_down(s2, off, 32);
  }
  if ((threadIdx.x & 31) == 0) { red[(threadIdx.x >> 5) * 2] = s; red[(threadIdx.x >> 5) * 2 + 1] = s2; }
  __syncthreads();
  if (threadIdx.x == 0) {
    float a = 0.f, c2 = 0.f;
    for (int w = 0; w < 8; ++w) { a += red[2 * w]; c2 += red[2 * w + 1]; }
    red[0] = a / (float)E_; red[1] = c2 / (float)E_;
  }
  __syncthreads();
  float mu = red[0];
  float rs = rsqrtf(red[1] - mu * mu + 1e-5f);
#pragma unroll
  for (int i = 0; i < 3; ++i) {
    int c = threadIdx.x + i * 256;
    h2[(size_t)row * E_ + c] = (bf16_t)((v[i] - mu) * rs * g[c] + bb_[c]);
  }
}

// --------------------------------------- MLP GEMM1: mid = gelu(h2 @ W1 + b1)
// macro-tiles: 512 (M) x 96 (N) -> 49152 wave jobs, 6144 blocks
__global__ __launch_bounds__(256) void k_gemm_gelu(const bf16_t* __restrict__ A,
                                                   const bf16_t* __restrict__ Bt,
                                                   const float* __restrict__ bias,
                                                   bf16_t* __restrict__ C) {
  int wid = blockIdx.x * 8 + (threadIdx.x >> 5), lane = threadIdx.x & 31;
  int mt = wid / 96, nt = wid % 96;
  int m0 = mt * 32, n0 = nt * 32;
  v8f acc[4] = {};
  gemm_2x2(A, E_, Bt, E_, E_, m0, n0, lane, acc);
#pragma unroll
  for (int j = 0; j < 2; ++j) {
    int n = n0 + j * 16 + (lane & 15);
    float bv = bias[n];
#pragma unroll
    for (int i = 0; i < 2; ++i) {
      int mb = m0 + i * 16 + ((lane >> 4) << 3);
      v8f a = acc[i * 2 + j];
#pragma unroll
      for (int r = 0; r < 8; ++r) {
        float v = a[r] + bv;
        float gl = 0.5f * v * (1.f + erff(v * 0.70710678118f));  // exact gelu
        C[(size_t)(mb + r) * HM_ + n] = (bf16_t)gl;
      }
    }
  }
}

// --------------------------- MLP GEMM2: out(x1) += mid @ W2 + b2  (in place)
// macro-tiles: 512 (M) x 24 (N) -> 12288 wave jobs, 1536 blocks
__global__ __launch_bounds__(256) void k_gemm_out(const bf16_t* __restrict__ A,
                                                  const bf16_t* __restrict__ Bt,
                                                  const float* __restrict__ bias,
                                                  float* __restrict__ C) {
  int wid = blockIdx.x * 8 + (threadIdx.x >> 5), lane = threadIdx.x & 31;
  int mt = wid / 24, nt = wid % 24;
  int m0 = mt * 32, n0 = nt * 32;
  v8f acc[4] = {};
  gemm_2x2(A, HM_, Bt, HM_, HM_, m0, n0, lane, acc);
#pragma unroll
  for (int j = 0; j < 2; ++j) {
    int n = n0 + j * 16 + (lane & 15);
    float bv = bias[n];
#pragma unroll
    for (int i = 0; i < 2; ++i) {
      int mb = m0 + i * 16 + ((lane >> 4) << 3);
      v8f a = acc[i * 2 + j];
#pragma unroll
      for (int r = 0; r < 8; ++r) {
        size_t idx = (size_t)(mb + r) * E_ + n;
        C[idx] = C[idx] + a[r] + bv;              // residual x1 already in C
      }
    }
  }
}

// ============================================================================
extern "C" void kernel_launch(void* const* d_in, const int* in_sizes, int n_in,
                              void* d_out, int out_size, void* d_ws, size_t ws_size,
                              hipStream_t stream) {
  const float* x     = (const float*)d_in[0];
  const float* hx    = (const float*)d_in[1];
  const float* ln1g  = (const float*)d_in[2];
  const float* ln1b  = (const float*)d_in[3];
  const float* ln2g  = (const float*)d_in[4];
  const float* ln2b  = (const float*)d_in[5];
  const float* Wbb   = (const float*)d_in[6];
  const float* bbb   = (const float*)d_in[7];
  const float* Wff1  = (const float*)d_in[8];
  const float* bff1  = (const float*)d_in[9];
  const float* Wff2  = (const float*)d_in[10];
  const float* bff2  = (const float*)d_in[11];
  const float* Wta   = (const float*)d_in[12];
  const float* bta   = (const float*)d_in[13];
  const float* Wtb   = (const float*)d_in[14];
  const float* btb   = (const float*)d_in[15];
  const float* thr   = (const float*)d_in[16];
  const float* leak  = (const float*)d_in[17];
  const float* steep = (const float*)d_in[18];
  const float* W1    = (const float*)d_in[19];
  const float* b1    = (const float*)d_in[20];
  const float* W2    = (const float*)d_in[21];
  const float* b2    = (const float*)d_in[22];
  float* out = (float*)d_out;

  char* ws = (char*)d_ws;
  size_t off = 0;
  auto alloc = [&](size_t bytes) -> char* {
    char* p = ws + off;
    off = (off + bytes + 255) & ~(size_t)255;
    return p;
  };
  bf16_t* normed  = (bf16_t*)alloc((size_t)ROWS * E_ * 2);
  float*  bbx     = (float*) alloc((size_t)ROWS * BB_ * 4);
  bf16_t* WbbxT   = (bf16_t*)alloc((size_t)BB_ * E_ * 2);
  bf16_t* WbbhT   = (bf16_t*)alloc((size_t)BB_ * U_ * 2);
  bf16_t* WheadsT = (bf16_t*)alloc((size_t)4 * U_ * BB_ * 2);
  bf16_t* W1T     = (bf16_t*)alloc((size_t)HM_ * E_ * 2);
  bf16_t* W2T     = (bf16_t*)alloc((size_t)E_ * HM_ * 2);
  bf16_t* cfc     = (bf16_t*)alloc((size_t)ROWS * E_ * 2);
  bf16_t* h2      = (bf16_t*)alloc((size_t)ROWS * E_ * 2);
  bf16_t* mid     = (bf16_t*)alloc((size_t)ROWS * HM_ * 2);

  // 1) weight conversion (f32 -> bf16, transposed to N-major for B fragments)
#define TRN(src, dst, K, N) \
  k_transpose<<<dim3(((K) * (N) + 255) / 256), dim3(256), 0, stream>>>(src, dst, K, N)
  TRN(Wbb,              WbbxT,                768,  512);   // x-part rows 0..767
  TRN(Wbb + 768 * 512,  WbbhT,                1024, 512);   // h-part rows 768..1791
  TRN(Wff1, WheadsT + 0 * (U_ * BB_),         512, 1024);
  TRN(Wff2, WheadsT + 1 * (U_ * BB_),         512, 1024);
  TRN(Wta,  WheadsT + 2 * (U_ * BB_),         512, 1024);
  TRN(Wtb,  WheadsT + 3 * (U_ * BB_),         512, 1024);
  TRN(W1,   W1T,                              768, 3072);
  TRN(W2,   W2T,                              3072, 768);
#undef TRN

  // 2) LayerNorm1 -> bf16
  k_ln1<<<dim3(ROWS), dim3(256), 0, stream>>>(x, ln1g, ln1b, normed);

  // 3) precompute x-part of backbone for all timesteps (parallel WMMA GEMM)
  k_gemm_bbx<<<dim3(1024), dim3(256), 0, stream>>>(normed, WbbxT, bbb, bbx);

  // 4) sequential CfC scan (single persistent workgroup, WMMA M=16 padded)
  k_scan<<<dim3(1), dim3(512), 0, stream>>>(hx, bbx, WbbhT, WheadsT,
                                            bff1, bff2, bta, btb,
                                            cfc, out + (size_t)ROWS * E_);

  // 5) LIF gate + residual -> d_out(x1); LayerNorm2 -> h2
  k_post<<<dim3(ROWS), dim3(256), 0, stream>>>(x, cfc, thr, leak, steep,
                                               ln2g, ln2b, out, h2);

  // 6) MLP
  k_gemm_gelu<<<dim3(6144), dim3(256), 0, stream>>>(h2, W1T, b1, mid);
  k_gemm_out<<<dim3(1536), dim3(256), 0, stream>>>(mid, W2T, b2, out);
}
